// BiLSTM_77403900609242
// MI455X (gfx1250) — matrix operand order repaired and checked by reading, not compiled, not run on hardware
//
#include <hip/hip_runtime.h>
#include <hip/hip_bf16.h>

#define T_SEQ 4096
#define E_IN  640          // 512 emb + 128 dummy
#define H_DIM 1024
#define G4    4096         // 4*H
#define TAGS  37
#define NB_REC 32          // persistent blocks for the recurrent scan

typedef __attribute__((ext_vector_type(16))) __bf16 v16bf;
typedef __attribute__((ext_vector_type(8)))  __bf16 v8bf;
typedef __attribute__((ext_vector_type(8)))  float  v8f;

union BF16Frag { v16bf v; v8bf h[2]; };

// ------------------------------------------------------------ TDM support

#if __has_builtin(__builtin_amdgcn_tensor_load_to_lds) && __has_builtin(__builtin_amdgcn_s_wait_tensorcnt)
#define HAVE_TDM 1
typedef __attribute__((ext_vector_type(4))) unsigned int tdm_u32x4;
typedef __attribute__((ext_vector_type(8))) int          tdm_i32x8;
typedef __attribute__((ext_vector_type(4))) int          tdm_i32x4;

// Issue a 2-D TENSOR_LOAD_TO_LDS: tile_d1 rows of tile_d0 elements (bf16),
// row stride row_stride_elems, packed contiguously at lds_off.
__device__ __forceinline__ void tdm_load_2d(const void* gsrc, unsigned lds_off,
                                            unsigned tile_d0, unsigned tile_d1,
                                            unsigned tensor_d0, unsigned tensor_d1,
                                            unsigned row_stride_elems) {
    const unsigned long long ga = (unsigned long long)(uintptr_t)gsrc;
    tdm_u32x4 g0;
    g0[0] = 1u;                                                   // count=1, user descriptor
    g0[1] = lds_off;                                              // lds_addr (bytes)
    g0[2] = (unsigned)ga;                                         // global_addr[31:0]
    g0[3] = (unsigned)((ga >> 32) & 0x01FFFFFFull) | (2u << 30);  // addr[56:32] | type=2
    tdm_i32x8 g1;
    g1[0] = (int)(1u << 16);                                      // data_size=1 (2B), wg_mask=0
    g1[1] = (int)(tensor_d0 << 16);                               // tensor_dim0[15:0] -> bits 63:48
    g1[2] = (int)((tensor_d0 >> 16) | (tensor_d1 << 16));         // dim0 hi | dim1 lo
    g1[3] = (int)((tensor_d1 >> 16) | (tile_d0 << 16));           // dim1 hi | tile_dim0
    g1[4] = (int)(tile_d1 & 0xFFFFu);                             // tile_dim1 (tile_dim2=0)
    g1[5] = (int)row_stride_elems;                                // tensor_dim0_stride[31:0]
    g1[6] = 0;
    g1[7] = 0;
    tdm_i32x4 z4 = (tdm_i32x4)0;
#if __clang_major__ >= 23
    tdm_i32x8 z8 = (tdm_i32x8)0;
    __builtin_amdgcn_tensor_load_to_lds(g0, g1, z4, z4, z8, 0);
#else
    __builtin_amdgcn_tensor_load_to_lds(g0, g1, z4, z4, 0);
#endif
}
#else
#define HAVE_TDM 0
#endif

// generic shared pointer -> LDS byte offset (low 32 bits per LDS aperture rule)
#define LDS_OFF(p) ((unsigned)(uintptr_t)(p))

// ------------------------------------------------------------ utilities

__device__ __forceinline__ void grid_sync(unsigned* cnt, unsigned target) {
    __syncthreads();
    if (threadIdx.x == 0) {
        __threadfence();
        atomicAdd(cnt, 1u);
        while (__hip_atomic_load(cnt, __ATOMIC_RELAXED, __HIP_MEMORY_SCOPE_AGENT) < target) {
            __builtin_amdgcn_s_sleep(1);
        }
        __threadfence();
    }
    __syncthreads();
}

__device__ __forceinline__ float sigmoidf_fast(float x) {
    return 1.0f / (1.0f + __expf(-x));
}

// ------------------------------------------------------------ init / prep

__global__ void k_init(unsigned* cnt, const float* __restrict__ h0,
                       const float* __restrict__ c0,
                       __bf16* __restrict__ h_bf, float* __restrict__ c_f) {
    int i = blockIdx.x * blockDim.x + threadIdx.x;
    if (i == 0) *cnt = 0u;
    if (i < H_DIM) {
        h_bf[i] = (__bf16)h0[i];
        c_f[i]  = c0[i];
    }
}

__global__ void k_cvt_bf16(const float* __restrict__ src, __bf16* __restrict__ dst, int n) {
    for (int i = blockIdx.x * blockDim.x + threadIdx.x; i < n; i += gridDim.x * blockDim.x)
        dst[i] = (__bf16)src[i];
}

__global__ void k_embed(const int* __restrict__ sentence,
                        const float* __restrict__ emb,
                        const float* __restrict__ dummy,
                        __bf16* __restrict__ X) {
    const int t = blockIdx.x;
    const int w = sentence[t];
    __bf16* xr = X + (size_t)t * E_IN;
    const float* er = emb + (size_t)w * 512;
    for (int j = threadIdx.x; j < 512; j += blockDim.x) xr[j] = (__bf16)er[j];
    const float* dr = dummy + (size_t)t * 128;
    for (int j = threadIdx.x; j < 128; j += blockDim.x) xr[512 + j] = (__bf16)dr[j];
}

// ------------------------------------------------------------ phase 1 GEMM
// PG[T][4H] = X[T][640] * Wih^T + (b_ih + b_hh).
// All 8 waves of a block share the same time-tile tM, so the 16x640 A panel is
// TDM-staged into LDS once per block; B streams from L2.

__global__ void __launch_bounds__(256) k_pregemm(const __bf16* __restrict__ X,
                                                 const __bf16* __restrict__ W,
                                                 const float* __restrict__ b_ih,
                                                 const float* __restrict__ b_hh,
                                                 float* __restrict__ PG) {
    const int lane = threadIdx.x & 31;
    const int wave = threadIdx.x >> 5;
    const int tM = blockIdx.x >> 5;                    // time-tile, wave-invariant
    const int tN = (blockIdx.x * 8 + wave) & 255;      // gate-tile
    const int half = lane >> 4;
    const int l = lane & 15;

    __shared__ __bf16 ldsA[16 * E_IN];                 // 20 KB A panel

    const __bf16* gA = X + (size_t)(tM * 16) * E_IN;
#if HAVE_TDM
    if (threadIdx.x == 0)
        tdm_load_2d(gA, LDS_OFF(ldsA), E_IN, 16, E_IN, T_SEQ, E_IN);
    __builtin_amdgcn_s_wait_tensorcnt(0);
    __syncthreads();
#else
    {
        const unsigned long long* s = (const unsigned long long*)gA;
        unsigned long long* d = (unsigned long long*)ldsA;
        for (int j = threadIdx.x; j < 16 * E_IN / 4; j += 256) d[j] = s[j];
        __syncthreads();
    }
#endif

    const __bf16* Alds = ldsA + (size_t)l * E_IN;
    const __bf16* Brow = W + (size_t)(tN * 16 + l) * E_IN;

    v8f acc = {};
    for (int k = 0; k < E_IN; k += 32) {
        BF16Frag a, b;
        // A from LDS: lane l<16 -> M=l, K in {k..k+7, k+16..k+23}; hi lanes +8
        a.h[0] = *(const v8bf*)(Alds + k + half * 8);
        a.h[1] = *(const v8bf*)(Alds + k + 16 + half * 8);
        // B from global: lane l<16 -> N=l, K in {k..k+7, k+16..k+23}; hi lanes +8
        b.h[0] = *(const v8bf*)(Brow + k + half * 8);
        b.h[1] = *(const v8bf*)(Brow + k + 16 + half * 8);
        if (k + 32 < E_IN) __builtin_prefetch(Brow + k + 32, 0, 3);
        acc = __builtin_amdgcn_wmma_f32_16x16x32_bf16(false, a.v, false, b.v,
                                                      (short)0, acc, false, false);
    }

    const int n = tN * 16 + l;
    const float bias = b_ih[n] + b_hh[n];
    #pragma unroll
    for (int j = 0; j < 8; ++j) {
        const int m = j + half * 8;                    // D layout: VGPR j -> rows j / j+8
        PG[(size_t)(tM * 16 + m) * G4 + n] = acc[j] + bias;
    }
}

// ------------------------------------------------------------ phase 2 scan
// 32 persistent blocks x 8 waves = 256 waves; each wave owns a 16-wide gate
// chunk, 32 wmma ops per step over K=1024.  h (2 KB) is TDM-staged into LDS
// each step; w_hh (8 MB bf16) streams from L2.  Block 0 applies the LSTM
// nonlinearity + actor/critic head.

__global__ void __launch_bounds__(256) k_recurrent(
        const float* __restrict__ PG, const __bf16* __restrict__ Whh,
        float* __restrict__ gates, __bf16* __restrict__ h_bf,
        float* __restrict__ c_f,
        const float* __restrict__ w_actor, const float* __restrict__ b_actor,
        const float* __restrict__ w_critic, const float* __restrict__ b_critic,
        const int* __restrict__ keys, float* __restrict__ out, unsigned* cnt) {
    const int lane = threadIdx.x & 31;
    const int wave = threadIdx.x >> 5;
    const int gw = blockIdx.x * 8 + wave;       // 0..255
    const int nbase = gw * 16;
    const int half = lane >> 4;
    const int l = lane & 15;
    const __bf16* Brow = Whh + (size_t)(nbase + l) * H_DIM;

    __shared__ __bf16 h_lds[H_DIM];
    __shared__ float hc[2 * H_DIM];
    __shared__ float lg[TAGS + 1];

    for (int t = 0; t < T_SEQ; ++t) {
        // ---- stage h into LDS (per block, per step)
#if HAVE_TDM
        if (threadIdx.x == 0)
            tdm_load_2d(h_bf, LDS_OFF(h_lds), H_DIM, 1, H_DIM, 1, H_DIM);
        __builtin_amdgcn_s_wait_tensorcnt(0);
        __syncthreads();
#else
        for (int j = threadIdx.x; j < H_DIM / 4; j += 256)
            ((unsigned long long*)h_lds)[j] = ((const unsigned long long*)h_bf)[j];
        __syncthreads();
#endif

        // ---- gates[nbase..nbase+16) = PG[t] + Whh @ h  (WMMA, h broadcast in A)
        v8f acc = {};
        for (int k = 0; k < H_DIM; k += 32) {
            BF16Frag a, b;
            a.h[0] = *(const v8bf*)(h_lds + k + half * 8);
            a.h[1] = *(const v8bf*)(h_lds + k + 16 + half * 8);
            b.h[0] = *(const v8bf*)(Brow + k + half * 8);
            b.h[1] = *(const v8bf*)(Brow + k + 16 + half * 8);
            acc = __builtin_amdgcn_wmma_f32_16x16x32_bf16(false, a.v, false, b.v,
                                                          (short)0, acc, false, false);
        }
        if (half == 0)
            gates[nbase + l] = acc[0] + PG[(size_t)t * G4 + nbase + l];

        grid_sync(cnt, (unsigned)NB_REC * (2u * (unsigned)t + 1u));

        if (blockIdx.x == 0) {
            // LSTM cell update (torch gate order i,f,g,o)
            for (int j = threadIdx.x; j < H_DIM; j += 256) {
                const float gi = gates[j];
                const float gf = gates[H_DIM + j];
                const float gg = gates[2 * H_DIM + j];
                const float go = gates[3 * H_DIM + j];
                const float c2 = sigmoidf_fast(gf) * c_f[j] + sigmoidf_fast(gi) * tanhf(gg);
                const float h2 = sigmoidf_fast(go) * tanhf(c2);
                c_f[j] = c2;
                h_bf[j] = (__bf16)h2;
                hc[j] = h2;
                hc[H_DIM + j] = c2;
            }
            __syncthreads();
            // actor (37) + critic (1) heads on hc[2048]
            for (int o = wave; o < TAGS + 1; o += 8) {
                const float* wr = (o < TAGS) ? (w_actor + (size_t)o * 2 * H_DIM) : w_critic;
                float p = 0.f;
                for (int k = lane; k < 2 * H_DIM; k += 32) p += wr[k] * hc[k];
                for (int off = 16; off; off >>= 1) p += __shfl_down(p, off);
                if (lane == 0) lg[o] = p + ((o < TAGS) ? b_actor[o] : b_critic[0]);
            }
            __syncthreads();
            if (threadIdx.x == 0) {
                float mx = lg[0]; int ai = 0;
                for (int i = 1; i < TAGS; ++i) if (lg[i] > mx) { mx = lg[i]; ai = i; }
                float s = 0.f;
                for (int i = 0; i < TAGS; ++i) s += __expf(lg[i] - mx);
                const float lse = mx + __logf(s);
                const unsigned k0 = (unsigned)keys[2 * t];
                const unsigned k1 = (unsigned)keys[2 * t + 1];
                const unsigned mix = (k0 ^ (k1 * 2654435761u)) ^ (k1 >> 13);
                out[t]              = (float)ai;
                out[T_SEQ + t]      = lg[ai] - lse;
                out[2 * T_SEQ + t]  = (float)(mix % TAGS);
                out[3 * T_SEQ + t]  = lg[TAGS];
                float* hl = out + 4 * T_SEQ + (size_t)t * TAGS;
                for (int i = 0; i < TAGS; ++i) hl[i] = lg[i];
            }
            __syncthreads();
        }
        grid_sync(cnt, (unsigned)NB_REC * (2u * (unsigned)t + 2u));
    }
}

// ------------------------------------------------------------ launch

extern "C" void kernel_launch(void* const* d_in, const int* in_sizes, int n_in,
                              void* d_out, int out_size, void* d_ws, size_t ws_size,
                              hipStream_t stream) {
    const int*   sentence = (const int*)d_in[0];
    const float* dummy    = (const float*)d_in[1];
    const float* emb      = (const float*)d_in[2];
    const float* w_ih     = (const float*)d_in[3];
    const float* w_hh     = (const float*)d_in[4];
    const float* b_ih     = (const float*)d_in[5];
    const float* b_hh     = (const float*)d_in[6];
    const float* w_actor  = (const float*)d_in[7];
    const float* b_actor  = (const float*)d_in[8];
    const float* w_critic = (const float*)d_in[9];
    const float* b_critic = (const float*)d_in[10];
    const float* h0       = (const float*)d_in[11];
    const float* c0       = (const float*)d_in[12];
    const int*   keys     = (const int*)d_in[13];
    (void)in_sizes; (void)n_in; (void)out_size; (void)ws_size;

    char* ws = (char*)d_ws;
    size_t off = 0;
    auto alloc = [&](size_t bytes) -> void* {
        void* p = ws + off;
        off += (bytes + 255) & ~(size_t)255;
        return p;
    };

    unsigned* cnt   = (unsigned*)alloc(256);
    __bf16*   X     = (__bf16*)alloc((size_t)T_SEQ * E_IN * 2);
    __bf16*   WihB  = (__bf16*)alloc((size_t)G4 * E_IN * 2);
    __bf16*   WhhB  = (__bf16*)alloc((size_t)G4 * H_DIM * 2);
    float*    PG    = (float*)alloc((size_t)T_SEQ * G4 * 4);
    __bf16*   h_bf  = (__bf16*)alloc((size_t)H_DIM * 2);
    float*    c_f   = (float*)alloc((size_t)H_DIM * 4);
    float*    gates = (float*)alloc((size_t)G4 * 4);

    k_init<<<dim3((H_DIM + 255) / 256), dim3(256), 0, stream>>>(cnt, h0, c0, h_bf, c_f);
    k_cvt_bf16<<<dim3(2048), dim3(256), 0, stream>>>(w_ih, WihB, G4 * E_IN);
    k_cvt_bf16<<<dim3(2048), dim3(256), 0, stream>>>(w_hh, WhhB, G4 * H_DIM);
    k_embed<<<dim3(T_SEQ), dim3(256), 0, stream>>>(sentence, emb, dummy, X);
    k_pregemm<<<dim3(8192), dim3(256), 0, stream>>>(X, WihB, b_ih, b_hh, PG);
    k_recurrent<<<dim3(NB_REC), dim3(256), 0, stream>>>(PG, WhhB, gates, h_bf, c_f,
                                                        w_actor, b_actor, w_critic, b_critic,
                                                        keys, (float*)d_out, cnt);
}
